// BiAttention_77129022701924
// MI455X (gfx1250) — compile-verified, matrix operand order checked by reading, not run on hardware
//
#include <hip/hip_runtime.h>
#include <hip/hip_bf16.h>

// ---------------------------------------------------------------------------
// BiAttention trilinear similarity for MI455X (gfx1250, wave32, WMMA + TDM).
//   s[b,m,n] = (c[b,m,:]*cqw) . q[b,n,:]  +  c[b,m,:].cw  +  c[b,n,:].qw + bias
// B=8, L=2048, D=256.
//
// Pipeline:
//   1) rows kernel: s0/s1 row dot products.
//   2) convert kernel: one-shot f32 -> bf16 hi/lo split, row-interleaved:
//      Aint[row] = [256 hi | 256 lo] of (c*cqw); Bint likewise for q.
//      (33.6 MB in d_ws, L2-resident on the 192 MB L2.)
//   3) GEMM kernel: per 128x128 block, ONE tensor_load_to_lds per matrix
//      stages a 512x128 bf16 tile (row pitch padded 1024->1056 B) into LDS;
//      inner loop is ds_load_b128 (all-immediate offsets) + bf16x3
//      v_wmma_f32_16x16x32_bf16.
// ---------------------------------------------------------------------------

typedef __attribute__((ext_vector_type(16))) __bf16 v16bf;
typedef __attribute__((ext_vector_type(8)))  float  v8f;
typedef __attribute__((ext_vector_type(4)))  unsigned int v4ui;
typedef __attribute__((ext_vector_type(8)))  int v8i;
typedef __attribute__((ext_vector_type(4)))  int v4i;

#define B_ 8
#define L_ 2048
#define D_ 256
#define BLD_ ((size_t)B_ * (size_t)L_ * (size_t)D_)

#define ROW_E   512                 // interleaved row: 256 hi + 256 lo bf16
#define ROW_B   1056                // LDS row pitch: 1024 B + 32 B pad
#define TILE_B  (128 * ROW_B)       // one staged matrix tile: 135168 B
#define SMEM_B  (2 * TILE_B)        // 270336 B of LDS per workgroup
#define MT_B    (16 * ROW_B)        // 16 rows: 16896 B (fits DS immediate)

#if defined(__HIP_DEVICE_COMPILE__) && __has_builtin(__builtin_amdgcn_tensor_load_to_lds)
#define HAVE_TDM 1
#else
#define HAVE_TDM 0
#endif

// ---- helpers --------------------------------------------------------------

__device__ __forceinline__ void load16c(const float* __restrict__ p, float* v) {
  const float4 a0 = *(const float4*)(p);
  const float4 a1 = *(const float4*)(p + 4);
  const float4 a2 = *(const float4*)(p + 8);
  const float4 a3 = *(const float4*)(p + 12);
  v[0]=a0.x;  v[1]=a0.y;  v[2]=a0.z;  v[3]=a0.w;
  v[4]=a1.x;  v[5]=a1.y;  v[6]=a1.z;  v[7]=a1.w;
  v[8]=a2.x;  v[9]=a2.y;  v[10]=a2.z; v[11]=a2.w;
  v[12]=a3.x; v[13]=a3.y; v[14]=a3.z; v[15]=a3.w;
}

// 16 f32 as two runs of 8: [0..7] at p, [8..15] at p+16 (fragment layout)
__device__ __forceinline__ void load16r(const float* __restrict__ p, float* v) {
  const float4 a0 = *(const float4*)(p);
  const float4 a1 = *(const float4*)(p + 4);
  const float4 a2 = *(const float4*)(p + 16);
  const float4 a3 = *(const float4*)(p + 20);
  v[0]=a0.x;  v[1]=a0.y;  v[2]=a0.z;  v[3]=a0.w;
  v[4]=a1.x;  v[5]=a1.y;  v[6]=a1.z;  v[7]=a1.w;
  v[8]=a2.x;  v[9]=a2.y;  v[10]=a2.z; v[11]=a2.w;
  v[12]=a3.x; v[13]=a3.y; v[14]=a3.z; v[15]=a3.w;
}

__device__ __forceinline__ void split_pack16(const float* v, v16bf& hi, v16bf& lo) {
  union { v16bf bf; unsigned short u[16]; } H, Lo;
#pragma unroll
  for (int i = 0; i < 16; ++i) {
    const float x = v[i];
    const unsigned short h = (unsigned short)(__float_as_uint(x) >> 16);
    const float hf = __uint_as_float(((unsigned)h) << 16);
    const float r = x - hf;
    const unsigned short l = (unsigned short)(__float_as_uint(r) >> 16);
    H.u[i] = h; Lo.u[i] = l;
  }
  hi = H.bf; lo = Lo.bf;
}

__device__ __forceinline__ void split_store16(const float* v,
                                              unsigned short* __restrict__ hi,
                                              unsigned short* __restrict__ lo) {
  unsigned h[8], l[8];
#pragma unroll
  for (int j = 0; j < 8; ++j) {
    const float x0 = v[2*j], x1 = v[2*j+1];
    const unsigned u0 = __float_as_uint(x0), u1 = __float_as_uint(x1);
    h[j] = (u1 & 0xFFFF0000u) | (u0 >> 16);
    const float r0 = x0 - __uint_as_float(u0 & 0xFFFF0000u);
    const float r1 = x1 - __uint_as_float(u1 & 0xFFFF0000u);
    l[j] = (__float_as_uint(r1) & 0xFFFF0000u) | (__float_as_uint(r0) >> 16);
  }
  uint4* hp = (uint4*)hi; uint4* lp = (uint4*)lo;
  hp[0] = make_uint4(h[0], h[1], h[2], h[3]);
  hp[1] = make_uint4(h[4], h[5], h[6], h[7]);
  lp[0] = make_uint4(l[0], l[1], l[2], l[3]);
  lp[1] = make_uint4(l[4], l[5], l[6], l[7]);
}

// bf16 fragment: elements 0..7 at p, 8..15 at p+16 ushorts (two 16 B loads)
__device__ __forceinline__ v16bf ld_frag16(const unsigned short* p) {
  union { uint4 u[2]; v16bf v; } t;
  t.u[0] = *(const uint4*)(p);
  t.u[1] = *(const uint4*)(p + 16);
  return t.v;
}

// ---------------------------------------------------------------------------
// s0[b*L+i] = c[b,i,:].c_weight ; s1[b*L+i] = c[b,i,:].q_weight
// ---------------------------------------------------------------------------
__global__ __launch_bounds__(256)
void biattn_rows_kernel(const float* __restrict__ c,
                        const float* __restrict__ cw,
                        const float* __restrict__ qw,
                        float* __restrict__ s0, float* __restrict__ s1) {
  const int i = blockIdx.x * blockDim.x + threadIdx.x;
  const float* p = c + (size_t)i * D_;
  float a0 = 0.f, a1 = 0.f;
#pragma unroll 4
  for (int k = 0; k < D_; k += 4) {
    const float4 v  = *(const float4*)(p + k);
    const float4 w0 = *(const float4*)(cw + k);
    const float4 w1 = *(const float4*)(qw + k);
    a0 += v.x*w0.x + v.y*w0.y + v.z*w0.z + v.w*w0.w;
    a1 += v.x*w1.x + v.y*w1.y + v.z*w1.z + v.w*w1.w;
  }
  s0[i] = a0; s1[i] = a1;
}

// ---------------------------------------------------------------------------
// One-shot conversion, row-interleaved hi/lo:
//   Aint[row*512 + k]       = bf16_hi((c*cqw)[row][k])
//   Aint[row*512 + 256 + k] = bf16_lo((c*cqw)[row][k])     (Bint likewise, q)
// ---------------------------------------------------------------------------
__global__ __launch_bounds__(256)
void biattn_convert_kernel(const float* __restrict__ c, const float* __restrict__ q,
                           const float* __restrict__ cqw,
                           unsigned short* __restrict__ Aint,
                           unsigned short* __restrict__ Bint) {
  const size_t t = (size_t)blockIdx.x * blockDim.x + threadIdx.x;
  const size_t e = t * 16;                 // 16 contiguous source elements
  const size_t row = e >> 8;               // D_ = 256
  const int k = (int)(e & (size_t)(D_ - 1));
  float a[16], bb[16], w[16];
  load16c(c + e, a);
  load16c(q + e, bb);
  load16c(cqw + k, w);
#pragma unroll
  for (int i = 0; i < 16; ++i) a[i] *= w[i];
  unsigned short* da = Aint + row * ROW_E + k;
  unsigned short* db = Bint + row * ROW_E + k;
  split_store16(a,  da, da + 256);
  split_store16(bb, db, db + 256);
}

// ---------------------------------------------------------------------------
// GEMM: one tensor_load_to_lds per matrix (512x128 bf16 tile, padded pitch
// 1056 B), then ds_load_b128 fragments (base VGPR + immediate offsets) +
// v_wmma_f32_16x16x32_bf16 (bf16x3). Block 128x128, 8 waves, wave tile 64x32.
// ---------------------------------------------------------------------------
__global__ __launch_bounds__(256)
void biattn_s2_lds_kernel(const unsigned short* __restrict__ Aint,
                          const unsigned short* __restrict__ Bint,
                          const float* __restrict__ bias,
                          const float* __restrict__ s0v, const float* __restrict__ s1v,
                          float* __restrict__ out) {
  extern __shared__ char smem[];           // [A tile 135168 B][B tile 135168 B]
  const int b     = blockIdx.z;
  const int tileM = blockIdx.y;
  const int tileN = blockIdx.x;
  const int wid   = threadIdx.x >> 5;
  const int lane  = threadIdx.x & 31;
  const int half  = lane >> 4;
  const int l16   = lane & 15;
  const int waveM = wid & 1;
  const int waveN = wid >> 1;
  const int mBase = tileM * 128 + waveM * 64;
  const int nBase = tileN * 128 + waveN * 32;

  const unsigned short* srcs[2] = {
    Aint + (size_t)(b * L_ + tileM * 128) * ROW_E,
    Bint + (size_t)(b * L_ + tileN * 128) * ROW_E
  };

#if HAVE_TDM
  // ---- Tensor Data Mover: one tensor_load_to_lds per 128 KB tile ---------
  if (wid == 0) {
    // D# group1: data_size=2B (1<<16), pad_enable (1<<20),
    // pad_interval=7 (256 DW = 1024 B rows), pad_amount=7 (8 DW = 32 B).
    v8i g1;
    g1[0] = (int)((1u << 16) | (1u << 20) | (7u << 22) | (7u << 25));
    g1[1] = (int)(512u << 16);      // tensor_dim0[15:0] = 512 elements/row
    g1[2] = (int)(16384u << 16);    // tensor_dim1[15:0] = 16384 rows
    g1[3] = (int)(512u << 16);      // tile_dim0 = 512
    g1[4] = 128;                    // tile_dim1 = 128 rows
    g1[5] = 512;                    // tensor_dim0_stride = 512 elements
    g1[6] = 0; g1[7] = 0;
    const v4i gz = { 0, 0, 0, 0 };
#pragma unroll
    for (int t = 0; t < 2; ++t) {
      const unsigned long long ga = (unsigned long long)(size_t)(srcs[t]);
      v4ui g0;
      g0[0] = 1u;                               // count=1, user descriptor
      g0[1] = (unsigned)(t * TILE_B);           // lds_addr (bytes)
      g0[2] = (unsigned)(ga & 0xFFFFFFFFu);     // global_addr[31:0]
      g0[3] = (unsigned)((ga >> 32) & 0x01FFFFFFu) | (2u << 30);  // hi | type=2
#if __clang_major__ >= 23
      const v8i gz8 = { 0, 0, 0, 0, 0, 0, 0, 0 };
      __builtin_amdgcn_tensor_load_to_lds(g0, g1, gz, gz, gz8, 0);
#else
      __builtin_amdgcn_tensor_load_to_lds(g0, g1, gz, gz, 0);
#endif
    }
    __builtin_amdgcn_s_wait_tensorcnt(0);
  }
  __syncthreads();
#else
  // ---- cooperative staging fallback (same padded LDS layout) -------------
  {
    const int tid = threadIdx.x;
#pragma unroll
    for (int t = 0; t < 2; ++t) {
      const unsigned short* src = srcs[t];
      char* dst = smem + t * TILE_B;
      for (int i = tid; i < 128 * 64; i += 256) {
        const int row = i >> 6, seg = i & 63;          // 64 x 16 B per row
        const uint4 v = *(const uint4*)(src + (size_t)row * ROW_E + seg * 8);
        *(uint4*)(dst + row * ROW_B + seg * 16) = v;
      }
    }
  }
  __syncthreads();
#endif

  // ---- per-thread fragment base addresses (everything else is immediate) --
  const char* pA = smem + (waveM * 64 + l16) * ROW_B + half * 16;
  const char* pB = smem + TILE_B + (waveN * 32 + l16) * ROW_B + half * 16;

  const v8f zero = {};
  v8f acc[4][2];
#pragma unroll
  for (int mt = 0; mt < 4; ++mt)
#pragma unroll
    for (int nt = 0; nt < 2; ++nt)
      acc[mt][nt] = zero;

#pragma unroll
  for (int kc = 0; kc < D_; kc += 32) {
    const int ko = kc * 2;   // byte offset along K within the hi/lo half-row
    v16bf bhi[2], blo[2];
#pragma unroll
    for (int nt = 0; nt < 2; ++nt) {
      bhi[nt] = ld_frag16((const unsigned short*)(pB + nt * MT_B + ko));
      blo[nt] = ld_frag16((const unsigned short*)(pB + nt * MT_B + 512 + ko));
    }
#pragma unroll
    for (int mt = 0; mt < 4; ++mt) {
      const v16bf ahi = ld_frag16((const unsigned short*)(pA + mt * MT_B + ko));
      const v16bf alo = ld_frag16((const unsigned short*)(pA + mt * MT_B + 512 + ko));
#pragma unroll
      for (int nt = 0; nt < 2; ++nt) {
        // bf16x3: (Ah+Al)(Bh+Bl) ~= AhBh + AhBl + AlBh
        acc[mt][nt] = __builtin_amdgcn_wmma_f32_16x16x32_bf16(
            false, ahi, false, bhi[nt], (short)0, acc[mt][nt], false, false);
        acc[mt][nt] = __builtin_amdgcn_wmma_f32_16x16x32_bf16(
            false, ahi, false, blo[nt], (short)0, acc[mt][nt], false, false);
        acc[mt][nt] = __builtin_amdgcn_wmma_f32_16x16x32_bf16(
            false, alo, false, bhi[nt], (short)0, acc[mt][nt], false, false);
      }
    }
  }

  // ---- epilogue: s2 + s0[m] + s1[n] + bias -------------------------------
  const float bv = bias[0];
#pragma unroll
  for (int mt = 0; mt < 4; ++mt) {
#pragma unroll
    for (int nt = 0; nt < 2; ++nt) {
      const int n = nBase + nt * 16 + l16;
      const float s1n = s1v[b * L_ + n];
#pragma unroll
      for (int r = 0; r < 8; ++r) {
        const int m = mBase + mt * 16 + half * 8 + r;
        out[((size_t)b * L_ + m) * (size_t)L_ + n] =
            acc[mt][nt][r] + s0v[b * L_ + m] + s1n + bv;
      }
    }
  }
}

// ---------------------------------------------------------------------------
// FALLBACK GEMM (ws too small): fused in-wave conversion (round-1 kernel).
// ---------------------------------------------------------------------------
__global__ __launch_bounds__(256)
void biattn_s2_fused_kernel(const float* __restrict__ c, const float* __restrict__ q,
                            const float* __restrict__ cqw, const float* __restrict__ bias,
                            const float* __restrict__ s0v, const float* __restrict__ s1v,
                            float* __restrict__ out) {
  const int b     = blockIdx.z;
  const int tileM = blockIdx.y;
  const int tileN = blockIdx.x;
  const int wid   = threadIdx.x >> 5;
  const int lane  = threadIdx.x & 31;
  const int half  = lane >> 4;
  const int l16   = lane & 15;
  const int waveM = wid & 1;
  const int waveN = wid >> 1;
  const int mBase = tileM * 128 + waveM * 64;
  const int nBase = tileN * 128 + waveN * 32;

  const float* cB = c + (size_t)b * L_ * D_;
  const float* qB = q + (size_t)b * L_ * D_;

  const v8f zero = {};
  v8f acc[4][2];
#pragma unroll
  for (int mt = 0; mt < 4; ++mt)
#pragma unroll
    for (int nt = 0; nt < 2; ++nt)
      acc[mt][nt] = zero;

  for (int kc = 0; kc < D_; kc += 32) {
    const int kb = kc + half * 8;
    float sc[16];
    load16r(cqw + kb, sc);
    v16bf bhi[2], blo[2];
#pragma unroll
    for (int nt = 0; nt < 2; ++nt) {
      float v[16];
      load16r(qB + (size_t)(nBase + nt * 16 + l16) * D_ + kb, v);
      split_pack16(v, bhi[nt], blo[nt]);
    }
#pragma unroll
    for (int mt = 0; mt < 4; ++mt) {
      float v[16];
      load16r(cB + (size_t)(mBase + mt * 16 + l16) * D_ + kb, v);
#pragma unroll
      for (int i = 0; i < 16; ++i) v[i] *= sc[i];
      v16bf ahi, alo;
      split_pack16(v, ahi, alo);
#pragma unroll
      for (int nt = 0; nt < 2; ++nt) {
        acc[mt][nt] = __builtin_amdgcn_wmma_f32_16x16x32_bf16(
            false, ahi, false, bhi[nt], (short)0, acc[mt][nt], false, false);
        acc[mt][nt] = __builtin_amdgcn_wmma_f32_16x16x32_bf16(
            false, ahi, false, blo[nt], (short)0, acc[mt][nt], false, false);
        acc[mt][nt] = __builtin_amdgcn_wmma_f32_16x16x32_bf16(
            false, alo, false, bhi[nt], (short)0, acc[mt][nt], false, false);
      }
    }
  }

  const float bv = bias[0];
#pragma unroll
  for (int mt = 0; mt < 4; ++mt) {
#pragma unroll
    for (int nt = 0; nt < 2; ++nt) {
      const int n = nBase + nt * 16 + l16;
      const float s1n = s1v[b * L_ + n];
#pragma unroll
      for (int r = 0; r < 8; ++r) {
        const int m = mBase + mt * 16 + half * 8 + r;
        out[((size_t)b * L_ + m) * (size_t)L_ + n] =
            acc[mt][nt][r] + s0v[b * L_ + m] + s1n + bv;
      }
    }
  }
}

// ---------------------------------------------------------------------------
extern "C" void kernel_launch(void* const* d_in, const int* in_sizes, int n_in,
                              void* d_out, int out_size, void* d_ws, size_t ws_size,
                              hipStream_t stream) {
  (void)in_sizes; (void)n_in; (void)out_size;
  const float* c    = (const float*)d_in[0];
  const float* q    = (const float*)d_in[1];
  const float* cw   = (const float*)d_in[2];
  const float* qw   = (const float*)d_in[3];
  const float* cqw  = (const float*)d_in[4];
  const float* bias = (const float*)d_in[5];
  float* out = (float*)d_out;

  // ws layout: s0 | s1 | Aint (2*BLD ushort) | Bint (2*BLD ushort)
  float* s0 = (float*)d_ws;
  float* s1 = s0 + B_ * L_;
  unsigned short* Aint = (unsigned short*)(s1 + B_ * L_);
  unsigned short* Bint = Aint + 2 * BLD_;
  const size_t needed = 2 * (size_t)B_ * L_ * sizeof(float)
                      + 4 * BLD_ * sizeof(unsigned short);

  biattn_rows_kernel<<<(B_ * L_) / 256, 256, 0, stream>>>(c, cw, qw, s0, s1);

  dim3 grid(L_ / 128, L_ / 128, B_);
  if (ws_size >= needed) {
    biattn_convert_kernel<<<(unsigned)(BLD_ / 16 / 256), 256, 0, stream>>>(
        c, q, cqw, Aint, Bint);
    biattn_s2_lds_kernel<<<grid, 256, SMEM_B, stream>>>(Aint, Bint,
                                                        bias, s0, s1, out);
  } else {
    biattn_s2_fused_kernel<<<grid, 256, 0, stream>>>(c, q, cqw, bias, s0, s1, out);
  }
}
